// TestDCN_83708912599177
// MI455X (gfx1250) — matrix operand order repaired and checked by reading, not compiled, not run on hardware
//
#include <hip/hip_runtime.h>

// ---- problem dims (fixed by reference) ----
#define B_   8
#define C_   128
#define H_   128
#define W_   128
#define OC_  256      // 2C
#define HW_  (H_ * W_)
#define HP_  130      // padded height (rows -1..128)
#define WPAD 130      // padded width  (cols -1..128)

// conv kernel dynamic-LDS partition
#define XS_BYTES   (3 * WPAD * 32 * 2)          // 24,960 B  : xs[3][130][32] bf16
#define WSM_ROWS   (9 * OC_)                    // 2304 rows : all taps x oc
#define WSM_BYTES  (WSM_ROWS * 32 * 2)          // 147,456 B : wsm[2304][32] bf16
#define SMEM_BYTES (XS_BYTES + WSM_BYTES)       // 172,416 B (< 320 KB/WGP)

typedef __attribute__((ext_vector_type(16))) __bf16 bf16x16;
typedef __attribute__((ext_vector_type(8)))  __bf16 bf16x8;
typedef __attribute__((ext_vector_type(8)))  float  f32x8;
typedef __attribute__((ext_vector_type(4)))  unsigned int u32x4;
typedef __attribute__((ext_vector_type(8)))  int i32x8;
typedef __attribute__((ext_vector_type(4)))  int i32x4;

union Frag16 { bf16x16 v; bf16x8 h[2]; };

// ---------------------------------------------------------------------------
// TDM descriptor build + issue (6-arg builtin on this toolchain).
// D# bit layout per CDNA5 ISA §8. 3-D tile: dim0 contiguous elements,
// dim1 lines (stride s0), dim2 planes (stride s1), in data_size(=2B) units.
// tile2==0 -> 2-D tile.
// ---------------------------------------------------------------------------
__device__ __forceinline__ void tdm_load_bf16(
    unsigned lds_byte_addr, const void* gptr,
    unsigned tile0, unsigned tile1, unsigned tile2,
    unsigned tdim0, unsigned tdim1, unsigned tdim2,
    unsigned long long s0, unsigned long long s1)
{
    unsigned long long ga = (unsigned long long)gptr;
    u32x4 g0 = { 1u,                                   // count=1 valid user D#
                 lds_byte_addr,                        // lds_addr
                 (unsigned)ga,                         // global_addr[31:0]
                 (unsigned)((ga >> 32) & 0x1FFFFFFull) // global_addr[56:32]
                   | (2u << 30) };                     // type=2 ("image")
    i32x8 g1 = { (int)0x00010000u,                               // mask=0, data_size=1 (2B)
                 (int)((tdim0 & 0xFFFFu) << 16),                 // tensor_dim0 lo16 @ bits48-63
                 (int)((tdim0 >> 16) | ((tdim1 & 0xFFFFu) << 16)),
                 (int)((tdim1 >> 16) | (tile0 << 16)),           // tile_dim0 @ bits112-127
                 (int)(tile1 | (tile2 << 16)),                   // tile_dim1, tile_dim2
                 (int)(s0 & 0xFFFFFFFFull),                      // tensor_dim0_stride lo32
                 (int)(((s0 >> 32) & 0xFFFFull) | ((s1 & 0xFFFFull) << 16)),
                 (int)((s1 >> 16) & 0xFFFFFFFFull) };            // tensor_dim1_stride hi32
    i32x4 g2 = { (int)tdim2, 0, 0, 0 };                          // tensor_dim2
    i32x4 g3 = { 0, 0, 0, 0 };
    i32x8 g4 = { 0, 0, 0, 0, 0, 0, 0, 0 };                       // extra slot (6-arg form)
    __builtin_amdgcn_tensor_load_to_lds(g0, g1, g2, g3, g4, 0);
}

// ---------------------------------------------------------------------------
// Prologue A: zero-fill the padded bf16 x tensor (stores as uint4).
// ---------------------------------------------------------------------------
__global__ __launch_bounds__(256)
void xp_zero(uint4* __restrict__ p)
{
    p[blockIdx.x * 256 + threadIdx.x] = make_uint4(0u, 0u, 0u, 0u);
}

// ---------------------------------------------------------------------------
// Prologue B: transpose/convert x (NCHW f32) -> xp[b][h+1][w+1][c] bf16.
// One block per 32x32 (c,w) tile; coalesced reads and writes via LDS tile.
// ---------------------------------------------------------------------------
__global__ __launch_bounds__(256)
void xp_transpose(const float* __restrict__ x, unsigned short* __restrict__ xp_)
{
    __bf16* xp = reinterpret_cast<__bf16*>(xp_);
    __shared__ float tile[32][33];
    const int bid = blockIdx.x;
    const int c0 = (bid & 3) * 32;
    const int w0 = ((bid >> 2) & 3) * 32;
    const int h  = (bid >> 4) & 127;
    const int b  = bid >> 11;
    const int tx = threadIdx.x & 31;
    const int ty = threadIdx.x >> 5;
    #pragma unroll
    for (int k = 0; k < 4; ++k) {
        int c = ty + k * 8;
        tile[c][tx] = x[((b * C_ + c0 + c) * H_ + h) * W_ + w0 + tx];
    }
    __syncthreads();
    #pragma unroll
    for (int k = 0; k < 4; ++k) {
        int w = ty + k * 8;
        long o = ((long)(b * HP_ + h + 1) * WPAD + (w0 + w + 1)) * C_ + c0 + tx;
        xp[o] = (__bf16)tile[tx][w];
    }
}

// ---------------------------------------------------------------------------
// Prologue C: W_off (OC,IC,3,3) f32 -> wt[tap][oc][ic] bf16 (ic contiguous).
// ---------------------------------------------------------------------------
__global__ __launch_bounds__(256)
void w_convert(const float* __restrict__ wsrc, unsigned short* __restrict__ wt_)
{
    __bf16* wt = reinterpret_cast<__bf16*>(wt_);
    int idx = blockIdx.x * 256 + threadIdx.x;  // [tap<<15 | oc<<7 | ic]
    int ic  = idx & 127;
    int oc  = (idx >> 7) & 255;
    int tap = idx >> 15;
    wt[idx] = (__bf16)wsrc[(oc * C_ + ic) * 9 + tap];
}

// ---------------------------------------------------------------------------
// Kernel 1: implicit-GEMM 3x3 conv on bf16 WMMA, LDS staged by TDM.
// Block = 256 thr (8 waves) computes one (b,h) row: 128 pixels x 256 oc.
// Per ic-chunk: ONE 3-D TDM for the x slab + ONE 2-D TDM for all 9 taps of
// weights (147 KB), then 144 WMMAs with no intervening barriers.
// ---------------------------------------------------------------------------
__global__ __launch_bounds__(256)
void dcn_offset_conv(const unsigned short* __restrict__ xp_,
                     const unsigned short* __restrict__ wt_,
                     float* __restrict__ offs)
{
    extern __shared__ __align__(16) char smem[];
    __bf16* xs  = reinterpret_cast<__bf16*>(smem);              // [3][WPAD][32]
    __bf16* wsm = reinterpret_cast<__bf16*>(smem + XS_BYTES);   // [2304][32]

    const __bf16* xp = reinterpret_cast<const __bf16*>(xp_);
    const __bf16* wt = reinterpret_cast<const __bf16*>(wt_);

    const int tid  = threadIdx.x;
    const int lane = tid & 31;
    const int wave = tid >> 5;
    const int b    = blockIdx.x / H_;
    const int h    = blockIdx.x % H_;

    const int mrow  = lane & 15;
    const int kbase = (lane >> 4) * 8;
    const int khalf = (lane >> 4) * 16;
    const int pix0  = wave * 16;

    const unsigned xs_lds  = (unsigned)(unsigned long long)xs;
    const unsigned wsm_lds = (unsigned)(unsigned long long)wsm;

    f32x8 acc[16] = {};

    for (int c0 = 0; c0 < C_; c0 += 32) {
        __syncthreads();                       // prior chunk's LDS readers done
        if (wave == 0) {
            // x slab: 3-D tile 32ch x 130cols x 3rows at padded (h,0,c0)
            const __bf16* gx = xp + ((long)(b * HP_ + h) * WPAD) * C_ + c0;
            tdm_load_bf16(xs_lds, gx, 32, WPAD, 3,
                          C_, WPAD, HP_,
                          (unsigned long long)C_,
                          (unsigned long long)(WPAD * C_));
            // weights: 2-D tile 32ch x (9 taps * 256 oc) rows, stride 128
            const __bf16* gw = wt + c0;
            tdm_load_bf16(wsm_lds, gw, 32, WSM_ROWS, 0,
                          C_, WSM_ROWS, 1,
                          (unsigned long long)C_,
                          (unsigned long long)(WSM_ROWS * C_));
            __builtin_amdgcn_s_wait_tensorcnt(0);
        }
        __syncthreads();                       // slabs ready for all waves

        for (int tap = 0; tap < 9; ++tap) {
            const int kh = tap / 3, kw = tap % 3;

            // A fragment: 16 pixels x 32 ic (ISA 16-bit A 16x32 lane map)
            Frag16 a;
            const __bf16* ap = xs + ((kh * WPAD + pix0 + mrow + kw) << 5);
            a.h[0] = *reinterpret_cast<const bf16x8*>(ap + kbase);
            a.h[1] = *reinterpret_cast<const bf16x8*>(ap + 16 + kbase);

            // B fragments: two-deep software pipeline over 16 N-tiles
            const __bf16* bbase = wsm + (((tap * OC_ + (lane & 15)) << 5) + kbase);
            Frag16 bfr[2];
            bfr[0].h[0] = *reinterpret_cast<const bf16x8*>(bbase);
            bfr[0].h[1] = *reinterpret_cast<const bf16x8*>(bbase + 16);
            #pragma unroll
            for (int n = 0; n < 16; ++n) {
                if (n < 15) {
                    const __bf16* bp = bbase + (n + 1) * (16 << 5);  // +16 oc rows
                    bfr[(n + 1) & 1].h[0] = *reinterpret_cast<const bf16x8*>(bp);
                    bfr[(n + 1) & 1].h[1] = *reinterpret_cast<const bf16x8*>(bp + 16);
                }
                acc[n] = __builtin_amdgcn_wmma_f32_16x16x32_bf16(
                             false, a.v, false, bfr[n & 1].v, (short)0, acc[n],
                             false, false);
            }
        }
    }

    // C/D layout: VGPR r -> M=r+8*(lane>=16), N=lane&15. 32B vector stores.
    const int oc_l = lane & 15;
    const int w0   = pix0 + 8 * (lane >> 4);
    #pragma unroll
    for (int n = 0; n < 16; ++n) {
        int obase = ((b * OC_ + n * 16 + oc_l) * H_ + h) * W_ + w0;
        *reinterpret_cast<f32x8*>(&offs[obase]) = acc[n];
    }
}

// ---------------------------------------------------------------------------
// Kernel 2: bilinear sampling with the reference's raw-reshape index math.
//   offsets (B,2C,H,W) viewed as (B*C, H*W, 2): flat = (b*C+c)*2*HW + 2q + t
// ---------------------------------------------------------------------------
__global__ __launch_bounds__(256)
void dcn_bilinear(const float* __restrict__ x,
                  const float* __restrict__ offs,
                  float* __restrict__ out)
{
    int idx = blockIdx.x * 256 + threadIdx.x;
    int q   = idx & (HW_ - 1);
    int pc  = idx >> 14;
    int base = pc * (2 * HW_) + 2 * q;

    float oy = offs[base + 0];
    float ox = offs[base + 1];
    float gy = (float)(q >> 7);
    float gx = (float)(q & (W_ - 1));
    float cy = fminf(fmaxf(gy + oy, 0.f), (float)(H_ - 1));
    float cx = fminf(fmaxf(gx + ox, 0.f), (float)(W_ - 1));
    float y0 = floorf(cy), y1 = ceilf(cy);
    float x0 = floorf(cx), x1 = ceilf(cx);
    int iy0 = (int)y0, iy1 = (int)y1, ix0 = (int)x0, ix1 = (int)x1;

    const float* xp = x + pc * HW_;
    float v_lt = xp[iy0 * W_ + ix0];
    float v_rb = xp[iy1 * W_ + ix1];
    float v_lb = xp[iy0 * W_ + ix1];
    float v_rt = xp[iy1 * W_ + ix0];

    float fy = cy - y0, fx = cx - x0;
    float v_t = v_lt + fy * (v_rt - v_lt);
    float v_b = v_lb + fy * (v_rb - v_lb);
    out[idx] = v_t + fx * (v_b - v_t);
}

// ---------------------------------------------------------------------------
extern "C" void kernel_launch(void* const* d_in, const int* in_sizes, int n_in,
                              void* d_out, int out_size, void* d_ws, size_t ws_size,
                              hipStream_t stream)
{
    const float* x     = (const float*)d_in[0];   // (8,128,128,128) f32
    const float* W_off = (const float*)d_in[1];   // (256,128,3,3)   f32
    float* out = (float*)d_out;

    // workspace layout
    //   [0, 128 MiB)          : offsets f32 (8,256,128,128)
    //   [+0x8000000, +33 MiB) : xp bf16 padded channel-last (8,130,130,128)
    //   then                  : wt bf16 (9,256,128)
    char* ws = (char*)d_ws;
    float*          offs = (float*)ws;
    unsigned short* xp   = (unsigned short*)(ws + 134217728);
    unsigned short* wt   = (unsigned short*)(ws + 134217728 + 34611200);

    // prologue: pad+transpose+convert inputs to bf16
    xp_zero<<<dim3(8450), dim3(256), 0, stream>>>((uint4*)xp);          // 34,611,200 B
    xp_transpose<<<dim3(16384), dim3(256), 0, stream>>>(x, xp);
    w_convert<<<dim3(1152), dim3(256), 0, stream>>>(W_off, wt);

    // conv via TDM-staged bf16 WMMA: one block per (b,h) row, 172 KB LDS
    dcn_offset_conv<<<dim3(B_ * H_), dim3(256), SMEM_BYTES, stream>>>(xp, wt, offs);

    // bilinear sampling: one thread per output element
    dcn_bilinear<<<dim3((B_ * C_ * HW_) / 256), dim3(256), 0, stream>>>(x, offs, out);
}
